// HSTUBlock_15375982920320
// MI455X (gfx1250) — matrix operand order repaired and checked by reading, not compiled, not run on hardware
//
#include <hip/hip_runtime.h>

// ---------- types ----------
typedef __bf16 bf16;
typedef bf16  bf16x16 __attribute__((ext_vector_type(16)));
typedef bf16  bf16x8  __attribute__((ext_vector_type(8)));
typedef float f32x8   __attribute__((ext_vector_type(8)));
typedef unsigned int u32x4 __attribute__((ext_vector_type(4)));
typedef int  i32x4v __attribute__((ext_vector_type(4)));
typedef int  i32x8v __attribute__((ext_vector_type(8)));

#define LN_EPS 1e-5f
#define Bx 4
#define Nx 1024
#define Dx 512
#define Hx 8
#define Ex 2048   // 2*H*Dv + 2*H*Dq

// Tensor Data Mover availability (device pass only; host pass takes fallback)
#if defined(__gfx1250__) && __has_builtin(__builtin_amdgcn_tensor_load_to_lds) && __has_builtin(__builtin_amdgcn_s_wait_tensorcnt)
#define HAVE_TDM 1
#else
#define HAVE_TDM 0
#endif

// ---------- WMMA helpers (CDNA5 gfx1250, wave32) ----------
__device__ __forceinline__ f32x8 wmma_bf16(bf16x16 a, bf16x16 b, f32x8 c) {
  // v_wmma_f32_16x16x32_bf16: D = A(16x32) * B(32x16) + C(16x16 f32)
  return __builtin_amdgcn_wmma_f32_16x16x32_bf16(false, a, false, b, (short)0, c, false, false);
}

// A-matrix 16x32 bf16 layout (ISA 7.12.2): lane (m = lane&15, half = lane>>4)
//   elems 0..7  = A[m][ half*8    .. +8 )
//   elems 8..15 = A[m][ 16+half*8 .. +8 )
__device__ __forceinline__ bf16x16 load_a_frag(const bf16* base, int ld) {
  int lane = threadIdx.x & 31;
  int m = lane & 15, half = lane >> 4;
  const bf16* row = base + (size_t)m * ld;
  bf16x8 c0 = *reinterpret_cast<const bf16x8*>(row + half * 8);
  bf16x8 c1 = *reinterpret_cast<const bf16x8*>(row + 16 + half * 8);
  bf16x16 r;
#pragma unroll
  for (int i = 0; i < 8; ++i) { r[i] = c0[i]; r[i + 8] = c1[i]; }
  return r;
}

// B-matrix 32x16 bf16 layout: lane (n = lane&15, half = lane>>4) holds
//   elems 0..7 = B[ half*8 .. +8 ][ n ], elems 8..15 = B[ 16+half*8 .. +8 ][ n ] (memory [n][k])
__device__ __forceinline__ bf16x16 load_b_frag(const bf16* base, int ld) {
  int lane = threadIdx.x & 31;
  int n = lane & 15, half = lane >> 4;
  const bf16* col = base + (size_t)n * ld;
  bf16x8 c0 = *reinterpret_cast<const bf16x8*>(col + half * 8);
  bf16x8 c1 = *reinterpret_cast<const bf16x8*>(col + 16 + half * 8);
  bf16x16 r;
#pragma unroll
  for (int i = 0; i < 8; ++i) { r[i] = c0[i]; r[i + 8] = c1[i]; }
  return r;
}

__device__ __forceinline__ float silu_f(float x) { return x / (1.0f + __expf(-x)); }

#if HAVE_TDM
// TDM: 2D tile load Global -> LDS.  D# per CDNA5 ISA.
//   group0: count=1 | lds_addr | global_addr[56:0] | type=2
//   group1: data_size=2B, pad 4 dwords every 128 dwords (LDS bank de-conflict),
//           tensor_dim0/1, tile_dim0/1, tensor_dim0_stride (elements)
__device__ __forceinline__ void tdm_load_2d(unsigned int lds_addr, const void* gptr,
                                            unsigned int tile_d0, unsigned int tile_d1,
                                            unsigned int tensor_d0, unsigned int tensor_d1,
                                            unsigned long long stride_e, unsigned int g1w0) {
  unsigned long long ga = (unsigned long long)(uintptr_t)gptr;
  u32x4 g0;
  g0[0] = 1u;                                   // count=1, user descriptor
  g0[1] = lds_addr;                             // LDS byte address
  g0[2] = (unsigned int)ga;                     // global_addr[31:0]
  g0[3] = (unsigned int)((ga >> 32) & 0x01FFFFFFu) | (2u << 30); // ga[56:32] | type=2
  i32x8v g1;
  g1[0] = (int)g1w0;                                             // mask=0|data_size|pad cfg
  g1[1] = (int)((tensor_d0 & 0xFFFFu) << 16);                    // tensor_dim0[15:0] @63:48
  g1[2] = (int)(((tensor_d0 >> 16) & 0xFFFFu) | ((tensor_d1 & 0xFFFFu) << 16));
  g1[3] = (int)(((tensor_d1 >> 16) & 0xFFFFu) | ((tile_d0 & 0xFFFFu) << 16)); // tile_dim0
  g1[4] = (int)(tile_d1 & 0xFFFFu);                              // tile_dim1 (tile_dim2=0)
  g1[5] = (int)(unsigned int)stride_e;                           // dim0 stride[31:0]
  g1[6] = (int)((stride_e >> 32) & 0xFFFFu);                     // dim0 stride[47:32]
  g1[7] = 0;
  i32x4v z4 = {0, 0, 0, 0};
#if __clang_major__ >= 23
  i32x8v z8 = {0, 0, 0, 0, 0, 0, 0, 0};
  __builtin_amdgcn_tensor_load_to_lds(g0, g1, z4, z4, z8, 0);
#else
  __builtin_amdgcn_tensor_load_to_lds(g0, g1, z4, z4, 0);
#endif
}
// data_size=2B | pad_enable | pad_interval=128dw(code 6) | pad_amount=4dw(code 3)
#define TDM_G1W0 ((1u << 16) | (1u << 20) | (6u << 22) | (3u << 25))
#define APAD 264   // 256 elements + 8 pad elements (132 dwords/row -> bank-friendly)
#endif

// ---------- K0: weight convert + transpose to bf16 (N-major for B-fragments) ----------
__global__ __launch_bounds__(256) void cvt_uvqk_kernel(const float* __restrict__ w, bf16* __restrict__ wt) {
  int idx = blockIdx.x * 256 + threadIdx.x;        // Ex*Dx threads
  int e = idx >> 9, d = idx & 511;                 // wt[e][d] = w[d][e]
  wt[idx] = (bf16)w[(size_t)d * Ex + e];
}
__global__ __launch_bounds__(256) void cvt_ow_kernel(const float* __restrict__ w, bf16* __restrict__ wt) {
  int idx = blockIdx.x * 256 + threadIdx.x;        // Dx*Dx threads
  int e = idx >> 9, d = idx & 511;                 // wt[e][d] = w[d][e]
  wt[idx] = (bf16)w[(size_t)d * Dx + e];
}

// ---------- K1: LayerNorm(x) -> nx (bf16), one wave per row ----------
__global__ __launch_bounds__(128) void ln_x_kernel(const float* __restrict__ x,
                                                   const float* __restrict__ w,
                                                   const float* __restrict__ b,
                                                   bf16* __restrict__ nx) {
  int row = blockIdx.x * 4 + (threadIdx.x >> 5);
  int lane = threadIdx.x & 31;
  const float* xr = x + (size_t)row * Dx;
  float v[16]; float s = 0.f;
#pragma unroll
  for (int i = 0; i < 16; ++i) { v[i] = xr[lane + i * 32]; s += v[i]; }
#pragma unroll
  for (int off = 16; off >= 1; off >>= 1) s += __shfl_xor(s, off, 32);
  float mean = s * (1.0f / Dx);
  float vs = 0.f;
#pragma unroll
  for (int i = 0; i < 16; ++i) { float d = v[i] - mean; vs += d * d; }
#pragma unroll
  for (int off = 16; off >= 1; off >>= 1) vs += __shfl_xor(vs, off, 32);
  float rstd = rsqrtf(vs * (1.0f / Dx) + LN_EPS);
#pragma unroll
  for (int i = 0; i < 16; ++i) {
    int c = lane + i * 32;
    nx[(size_t)row * Dx + c] = (bf16)((v[i] - mean) * rstd * w[c] + b[c]);
  }
}

// ---------- K2: projection GEMM  silu(nx @ uvqk)  -> u(f32) / Vt(bf16,T) / q(bf16) / k(bf16) ----------
// 2 waves/block; A panel (16x512) staged to LDS via TDM (2 halves, double-buffered);
// B fragments double-buffered in registers so WMMAs retire against older loads.
__global__ __launch_bounds__(64) void proj_kernel(const bf16* __restrict__ nx,
                                                  const bf16* __restrict__ wt,   // [Ex][Dx]
                                                  float* __restrict__ ub,        // [4096][512]
                                                  bf16* __restrict__ vt,         // [B*512][N] (dv-major)
                                                  bf16* __restrict__ qb,         // [4096][512]
                                                  bf16* __restrict__ kb) {       // [4096][512]
  int wv = threadIdx.x >> 5;
  int tile = blockIdx.x * 2 + wv;                  // 256*32 tiles
  int mt = tile >> 5, nt = tile & 31;
  int row0 = mt * 16, col0 = nt * 64;
  f32x8 acc[4] = {};

#if HAVE_TDM
  __shared__ __align__(1024) bf16 sA[2][2][16][APAD];
  {
    unsigned int l0 = (unsigned int)(uintptr_t)&sA[wv][0][0][0];
    unsigned int l1 = (unsigned int)(uintptr_t)&sA[wv][1][0][0];
    const bf16* g = nx + (size_t)row0 * Dx;
    tdm_load_2d(l0, g,       256, 16, 256, 16, 512, TDM_G1W0);
    tdm_load_2d(l1, g + 256, 256, 16, 256, 16, 512, TDM_G1W0);
  }
#endif

  bf16x16 bcur[4];
#pragma unroll
  for (int t = 0; t < 4; ++t)
    bcur[t] = load_b_frag(wt + (size_t)(col0 + t * 16) * Dx, Dx);

#if HAVE_TDM
  __builtin_amdgcn_s_wait_tensorcnt(1);            // first half-panel resident
#endif

#pragma unroll
  for (int k0 = 0; k0 < Dx; k0 += 32) {
#if HAVE_TDM
    if (k0 == 256) __builtin_amdgcn_s_wait_tensorcnt(0);
    bf16x16 a = load_a_frag(&sA[wv][k0 >> 8][0][0] + (k0 & 255), APAD);
#else
    bf16x16 a = load_a_frag(nx + (size_t)row0 * Dx + k0, Dx);
#endif
    bf16x16 bnxt[4];
    if (k0 + 32 < Dx) {
#pragma unroll
      for (int t = 0; t < 4; ++t)
        bnxt[t] = load_b_frag(wt + (size_t)(col0 + t * 16) * Dx + k0 + 32, Dx);
    }
#pragma unroll
    for (int t = 0; t < 4; ++t) acc[t] = wmma_bf16(a, bcur[t], acc[t]);
    if (k0 + 32 < Dx) {
#pragma unroll
      for (int t = 0; t < 4; ++t) bcur[t] = bnxt[t];
    }
  }

  int lane = threadIdx.x & 31;
  int nn = lane & 15, half = lane >> 4;
#pragma unroll
  for (int t = 0; t < 4; ++t) {
#pragma unroll
    for (int g = 0; g < 8; ++g) {
      float s = silu_f(acc[t][g]);
      int r = row0 + g + half * 8;
      int e = col0 + t * 16 + nn;
      if (e < 512) {                               // u
        ub[(size_t)r * 512 + e] = s;
      } else if (e < 1024) {                       // v -> transposed (dv-major)
        int bb = r >> 10, j = r & 1023;
        vt[(size_t)(bb * 512 + (e - 512)) * Nx + j] = (bf16)s;
      } else if (e < 1536) {                       // q
        qb[(size_t)r * 512 + (e - 1024)] = (bf16)s;
      } else {                                     // k
        kb[(size_t)r * 512 + (e - 1536)] = (bf16)s;
      }
    }
  }
}

// ---------- K3: attention  AO = (mask(silu(QK^T + relbias))/N) @ V ----------
__global__ __launch_bounds__(128) void attn_kernel(const bf16* __restrict__ qb,
                                                   const bf16* __restrict__ kb,
                                                   const bf16* __restrict__ vt,
                                                   const long long* __restrict__ ts,
                                                   const float* __restrict__ ts_w,
                                                   const float* __restrict__ pos_w,
                                                   const unsigned char* __restrict__ pm,
                                                   float* __restrict__ ao) {
  __shared__ __align__(64) bf16 sS[4][16][32];     // per-wave 16x32 score slab (bf16)
  int wv = threadIdx.x >> 5;
  int tile = blockIdx.x * 4 + wv;                  // 2048 tiles: b*H*64 + h*64 + it
  int it = tile & 63;
  int h = (tile >> 6) & 7;
  int b = tile >> 9;
  int i0 = it * 16;
  int lane = threadIdx.x & 31;
  int nn = lane & 15, half = lane >> 4;

  const bf16* Qbase = qb + (size_t)(b * Nx + i0) * 512 + h * 64;
  bf16x16 aq0 = load_a_frag(Qbase, 512);
  bf16x16 aq1 = load_a_frag(Qbase + 32, 512);

  f32x8 acc[4] = {};

  const long long* tsb = ts + (size_t)b * Nx;
  long long ts_ip1[8];
#pragma unroll
  for (int g = 0; g < 8; ++g) {
    int i = i0 + g + half * 8;
    ts_ip1[g] = tsb[i + 1 < Nx ? i + 1 : Nx - 1];  // ext[:,1:] term
  }

  const int imax = i0 + 15;
  for (int j0 = 0; j0 <= imax; j0 += 32) {
    // V fragments don't depend on scores: issue early, overlap with QK^T + postproc
    bf16x16 bv[4];
#pragma unroll
    for (int t = 0; t < 4; ++t)
      bv[t] = load_b_frag(vt + (size_t)(b * 512 + h * 64 + t * 16) * Nx + j0, Nx);
    // prefetch next K chunk into caches (global_prefetch_b8)
    if (j0 + 32 <= imax)
      __builtin_prefetch(kb + (size_t)(b * Nx + j0 + 32) * 512 + h * 64, 0, 0);

#pragma unroll
    for (int sub = 0; sub < 2; ++sub) {
      int jt0 = j0 + sub * 16;
      float vals[8] = {};
      if (jt0 <= imax) {                           // uniform branch: EXEC stays full
        const bf16* Kbase = kb + (size_t)(b * Nx + jt0) * 512 + h * 64;
        bf16x16 bk0 = load_b_frag(Kbase, 512);
        bf16x16 bk1 = load_b_frag(Kbase + 32, 512);
        f32x8 c = {};
        c = wmma_bf16(aq0, bk0, c);
        c = wmma_bf16(aq1, bk1, c);
        int j = jt0 + nn;
        long long tsj = tsb[j];
        bool pmj = pm[b * Nx + j] != 0;
#pragma unroll
        for (int g = 0; g < 8; ++g) {
          int i = i0 + g + half * 8;
          if (j <= i && !pmj) {                    // causal (triu k=1) + pad mask
            long long dll = ts_ip1[g] - tsj;
            float ad = fabsf((float)dll);
            float logd = __logf(fmaxf(ad, 1.0f));
            int bk = (int)(logd * (1.0f / 0.301f));
            bk = bk < 0 ? 0 : (bk > 64 ? 64 : bk);
            float bias = pos_w[j - i + (Nx - 1)] + ts_w[bk];
            vals[g] = silu_f(c[g] + bias) * (1.0f / Nx);
          }
        }
      }
#pragma unroll
      for (int g = 0; g < 8; ++g)
        sS[wv][g + half * 8][sub * 16 + nn] = (bf16)vals[g];
    }
    // intra-wave LDS store -> load ordering (CDNA5 split DS counter)
    asm volatile("s_wait_dscnt 0" ::: "memory");
    bf16x16 aS = load_a_frag(&sS[wv][0][0], 32);
#pragma unroll
    for (int t = 0; t < 4; ++t) acc[t] = wmma_bf16(aS, bv[t], acc[t]);
    asm volatile("s_wait_dscnt 0" ::: "memory");   // protect sS WAR before next chunk
  }
#pragma unroll
  for (int t = 0; t < 4; ++t)
#pragma unroll
    for (int g = 0; g < 8; ++g)
      ao[(size_t)(b * Nx + i0 + g + half * 8) * 512 + h * 64 + t * 16 + nn] = acc[t][g];
}

// ---------- K4: a2 = u * LayerNorm(ao)  (bf16) ----------
__global__ __launch_bounds__(128) void ln_a_kernel(const float* __restrict__ ao,
                                                   const float* __restrict__ u,
                                                   const float* __restrict__ w,
                                                   const float* __restrict__ b,
                                                   bf16* __restrict__ a2) {
  int row = blockIdx.x * 4 + (threadIdx.x >> 5);
  int lane = threadIdx.x & 31;
  const float* ar = ao + (size_t)row * 512;
  float v[16]; float s = 0.f;
#pragma unroll
  for (int i = 0; i < 16; ++i) { v[i] = ar[lane + i * 32]; s += v[i]; }
#pragma unroll
  for (int off = 16; off >= 1; off >>= 1) s += __shfl_xor(s, off, 32);
  float mean = s * (1.0f / 512.0f);
  float vs = 0.f;
#pragma unroll
  for (int i = 0; i < 16; ++i) { float d = v[i] - mean; vs += d * d; }
#pragma unroll
  for (int off = 16; off >= 1; off >>= 1) vs += __shfl_xor(vs, off, 32);
  float rstd = rsqrtf(vs * (1.0f / 512.0f) + LN_EPS);
#pragma unroll
  for (int i = 0; i < 16; ++i) {
    int c = lane + i * 32;
    float ln = (v[i] - mean) * rstd * w[c] + b[c];
    a2[(size_t)row * 512 + c] = (bf16)(ln * u[(size_t)row * 512 + c]);
  }
}

// ---------- K5: y = a2 @ o_w + o_b + x, masked by pm ----------
__global__ __launch_bounds__(64) void out_kernel(const bf16* __restrict__ a2,
                                                 const bf16* __restrict__ owt,  // [Dx][Dx] (col-major of o_w)
                                                 const float* __restrict__ ob,
                                                 const float* __restrict__ x,
                                                 const unsigned char* __restrict__ pm,
                                                 float* __restrict__ y) {
  int wv = threadIdx.x >> 5;
  int tile = blockIdx.x * 2 + wv;                  // 256*8 tiles
  int mt = tile >> 3, nt = tile & 7;
  int row0 = mt * 16, col0 = nt * 64;
  f32x8 acc[4] = {};

#if HAVE_TDM
  __shared__ __align__(1024) bf16 sA[2][2][16][APAD];
  {
    unsigned int l0 = (unsigned int)(uintptr_t)&sA[wv][0][0][0];
    unsigned int l1 = (unsigned int)(uintptr_t)&sA[wv][1][0][0];
    const bf16* g = a2 + (size_t)row0 * Dx;
    tdm_load_2d(l0, g,       256, 16, 256, 16, 512, TDM_G1W0);
    tdm_load_2d(l1, g + 256, 256, 16, 256, 16, 512, TDM_G1W0);
  }
#endif

  bf16x16 bcur[4];
#pragma unroll
  for (int t = 0; t < 4; ++t)
    bcur[t] = load_b_frag(owt + (size_t)(col0 + t * 16) * Dx, Dx);

#if HAVE_TDM
  __builtin_amdgcn_s_wait_tensorcnt(1);
#endif

#pragma unroll
  for (int k0 = 0; k0 < Dx; k0 += 32) {
#if HAVE_TDM
    if (k0 == 256) __builtin_amdgcn_s_wait_tensorcnt(0);
    bf16x16 a = load_a_frag(&sA[wv][k0 >> 8][0][0] + (k0 & 255), APAD);
#else
    bf16x16 a = load_a_frag(a2 + (size_t)row0 * Dx + k0, Dx);
#endif
    bf16x16 bnxt[4];
    if (k0 + 32 < Dx) {
#pragma unroll
      for (int t = 0; t < 4; ++t)
        bnxt[t] = load_b_frag(owt + (size_t)(col0 + t * 16) * Dx + k0 + 32, Dx);
    }
#pragma unroll
    for (int t = 0; t < 4; ++t) acc[t] = wmma_bf16(a, bcur[t], acc[t]);
    if (k0 + 32 < Dx) {
#pragma unroll
      for (int t = 0; t < 4; ++t) bcur[t] = bnxt[t];
    }
  }

  int lane = threadIdx.x & 31;
  int nn = lane & 15, half = lane >> 4;
#pragma unroll
  for (int t = 0; t < 4; ++t) {
#pragma unroll
    for (int g = 0; g < 8; ++g) {
      int r = row0 + g + half * 8;
      int c = col0 + t * 16 + nn;
      int bb = r >> 10, i = r & 1023;
      float v = pm[bb * Nx + i] ? 0.0f
                                : acc[t][g] + ob[c] + x[(size_t)r * Dx + c];
      y[(size_t)r * Dx + c] = v;
    }
  }
}

// ---------- launch ----------
extern "C" void kernel_launch(void* const* d_in, const int* in_sizes, int n_in,
                              void* d_out, int out_size, void* d_ws, size_t ws_size,
                              hipStream_t stream) {
  const float*          x      = (const float*)d_in[0];
  const long long*      ts     = (const long long*)d_in[1];
  /* cm (d_in[2]) is the static causal triu mask -> implemented analytically */
  const unsigned char*  pm     = (const unsigned char*)d_in[3];
  const float*          uvqk   = (const float*)d_in[4];
  const float*          o_w    = (const float*)d_in[5];
  const float*          o_b    = (const float*)d_in[6];
  const float*          ln_x_w = (const float*)d_in[7];
  const float*          ln_x_b = (const float*)d_in[8];
  const float*          ln_a_w = (const float*)d_in[9];
  const float*          ln_a_b = (const float*)d_in[10];
  const float*          ts_w   = (const float*)d_in[11];
  const float*          pos_w  = (const float*)d_in[12];
  float* y = (float*)d_out;

  const size_t ROWS = (size_t)Bx * Nx;             // 4096
  char* w = (char*)d_ws;
  size_t off = 0;
  auto alloc = [&](size_t bytes) { char* p = w + off; off = (off + bytes + 255) & ~(size_t)255; return p; };
  bf16*  nx  = (bf16*)alloc(ROWS * Dx * 2);        // 4 MiB
  bf16*  wt  = (bf16*)alloc((size_t)Ex * Dx * 2);  // 2 MiB
  bf16*  owt = (bf16*)alloc((size_t)Dx * Dx * 2);  // 0.5 MiB
  float* ub  = (float*)alloc(ROWS * 512 * 4);      // 8 MiB
  bf16*  vt  = (bf16*)alloc((size_t)Bx * 512 * Nx * 2); // 4 MiB
  bf16*  qb  = (bf16*)alloc(ROWS * 512 * 2);       // 4 MiB
  bf16*  kb  = (bf16*)alloc(ROWS * 512 * 2);       // 4 MiB
  float* ao  = (float*)alloc(ROWS * 512 * 4);      // 8 MiB
  bf16*  a2  = (bf16*)alloc(ROWS * 512 * 2);       // 4 MiB   (~38.5 MiB total)

  cvt_uvqk_kernel<<<(Ex * Dx) / 256, 256, 0, stream>>>(uvqk, wt);
  cvt_ow_kernel<<<(Dx * Dx) / 256, 256, 0, stream>>>(o_w, owt);
  ln_x_kernel<<<ROWS / 4, 128, 0, stream>>>(x, ln_x_w, ln_x_b, nx);
  proj_kernel<<<(256 * 32) / 2, 64, 0, stream>>>(nx, wt, ub, vt, qb, kb);
  attn_kernel<<<(Bx * Hx * 64) / 4, 128, 0, stream>>>(qb, kb, vt, ts, ts_w, pos_w, pm, ao);
  ln_a_kernel<<<ROWS / 4, 128, 0, stream>>>(ao, ub, ln_a_w, ln_a_b, a2);
  out_kernel<<<(256 * 8) / 2, 64, 0, stream>>>(a2, owt, o_b, x, pm, y);
}
